// Sage2_81527069213097
// MI455X (gfx1250) — matrix operand order repaired and checked by reading, not compiled, not run on hardware
//
#include <hip/hip_runtime.h>
#include <hip/hip_bf16.h>

typedef __bf16 bf16;
typedef __attribute__((ext_vector_type(16))) __bf16 v16bf;
typedef __attribute__((ext_vector_type(8)))  float  v8f;

// ---------------------------------------------------------------------------
// degree count (int atomics) + inverse
// ---------------------------------------------------------------------------
__global__ __launch_bounds__(256) void count_kernel(const int* __restrict__ rows,
                                                    int* __restrict__ cnt, int E) {
    int e = blockIdx.x * 256 + threadIdx.x;
    if (e < E) atomicAdd(cnt + rows[e], 1);
}

__global__ __launch_bounds__(256) void inv_kernel(const int* __restrict__ cnt,
                                                  float* __restrict__ inv, int N) {
    int i = blockIdx.x * 256 + threadIdx.x;
    if (i < N) inv[i] = 1.0f / ((float)cnt[i] + 1.0f);
}

// ---------------------------------------------------------------------------
// exclusive scan of cnt[N] -> rowstart[N]   (1024-element tiles)
// ---------------------------------------------------------------------------
__global__ __launch_bounds__(256) void scan_partial(const int* __restrict__ cnt,
                                                    int* __restrict__ blockSums,
                                                    int N) {
    __shared__ int s[256];
    const int t = threadIdx.x;
    const int base = blockIdx.x * 1024;
    int sum = 0;
    #pragma unroll
    for (int j = 0; j < 4; ++j) {
        int i = base + t * 4 + j;
        sum += (i < N) ? cnt[i] : 0;
    }
    s[t] = sum;
    __syncthreads();
    for (int off = 128; off > 0; off >>= 1) {
        if (t < off) s[t] += s[t + off];
        __syncthreads();
    }
    if (t == 0) blockSums[blockIdx.x] = s[0];
}

__global__ void scan_sums(int* __restrict__ blockSums, int nb) {   // <<<1,1>>>
    int acc = 0;
    for (int i = 0; i < nb; ++i) {
        int v = blockSums[i];
        blockSums[i] = acc;
        acc += v;
    }
}

__global__ __launch_bounds__(256) void scan_final(const int* __restrict__ cnt,
                                                  const int* __restrict__ blockSums,
                                                  int* __restrict__ rowstart, int N) {
    __shared__ int s[256];
    const int t = threadIdx.x;
    const int base = blockIdx.x * 1024;
    int v[4], sum = 0;
    #pragma unroll
    for (int j = 0; j < 4; ++j) {
        int i = base + t * 4 + j;
        v[j] = (i < N) ? cnt[i] : 0;
        sum += v[j];
    }
    s[t] = sum;
    __syncthreads();
    // inclusive Hillis-Steele over 256 thread sums
    for (int off = 1; off < 256; off <<= 1) {
        int add = (t >= off) ? s[t - off] : 0;
        __syncthreads();
        s[t] += add;
        __syncthreads();
    }
    int run = blockSums[blockIdx.x] + s[t] - sum;   // exclusive thread offset
    #pragma unroll
    for (int j = 0; j < 4; ++j) {
        int i = base + t * 4 + j;
        if (i < N) rowstart[i] = run;
        run += v[j];
    }
}

// ---------------------------------------------------------------------------
// CSR bucket fill: adj[rowstart[r] + cur[r]++] = col   (int atomics only)
// ---------------------------------------------------------------------------
__global__ __launch_bounds__(256) void fill_adj(const int* __restrict__ rows,
                                                const int* __restrict__ cols,
                                                const int* __restrict__ rowstart,
                                                int* __restrict__ cur,
                                                int* __restrict__ adj, int E) {
    int e = blockIdx.x * 256 + threadIdx.x;
    if (e < E) {
        int r = rows[e];
        int p = atomicAdd(cur + r, 1);
        adj[rowstart[r] + p] = cols[e];
    }
}

// ---------------------------------------------------------------------------
// Atomic-free gather aggregation: one wave32 per node.
//   dst[node, :] = inv[node] * sum_{c in adj(node)} src[c, :]
// Neighbor loop unrolled x4 so up to 4*V float4 gathers are in flight.
// ---------------------------------------------------------------------------
template <int F>
__global__ __launch_bounds__(256) void aggregate(const float* __restrict__ src,
                                                 const int* __restrict__ adj,
                                                 const int* __restrict__ rowstart,
                                                 const int* __restrict__ cnt,
                                                 const float* __restrict__ inv,
                                                 float* __restrict__ dst, int N) {
    const int node = blockIdx.x * 8 + (threadIdx.x >> 5);
    const int lane = threadIdx.x & 31;
    if (node >= N) return;
    const int start = rowstart[node];
    const int deg   = cnt[node];
    const float sc  = inv[node];
    constexpr int V = F / 128;                 // float4's per lane
    float4 acc[V];
    #pragma unroll
    for (int v = 0; v < V; ++v) acc[v] = make_float4(0.f, 0.f, 0.f, 0.f);

    int j = 0;
    for (; j + 4 <= deg; j += 4) {
        int c0 = adj[start + j + 0];
        int c1 = adj[start + j + 1];
        int c2 = adj[start + j + 2];
        int c3 = adj[start + j + 3];
        const float4* p0 = (const float4*)(src + (size_t)c0 * F);
        const float4* p1 = (const float4*)(src + (size_t)c1 * F);
        const float4* p2 = (const float4*)(src + (size_t)c2 * F);
        const float4* p3 = (const float4*)(src + (size_t)c3 * F);
        #pragma unroll
        for (int v = 0; v < V; ++v) {
            float4 a = p0[lane + v * 32];
            float4 b = p1[lane + v * 32];
            float4 c = p2[lane + v * 32];
            float4 d = p3[lane + v * 32];
            acc[v].x += a.x + b.x + c.x + d.x;
            acc[v].y += a.y + b.y + c.y + d.y;
            acc[v].z += a.z + b.z + c.z + d.z;
            acc[v].w += a.w + b.w + c.w + d.w;
        }
    }
    for (; j < deg; ++j) {
        const int c = adj[start + j];
        const float4* s4 = (const float4*)(src + (size_t)c * F);
        #pragma unroll
        for (int v = 0; v < V; ++v) {
            float4 xv = s4[lane + v * 32];
            acc[v].x += xv.x; acc[v].y += xv.y;
            acc[v].z += xv.z; acc[v].w += xv.w;
        }
    }
    float4* d4 = (float4*)(dst + (size_t)node * F);
    #pragma unroll
    for (int v = 0; v < V; ++v) {
        acc[v].x *= sc; acc[v].y *= sc; acc[v].z *= sc; acc[v].w *= sc;
        d4[lane + v * 32] = acc[v];
    }
}

// ---------------------------------------------------------------------------
// f32 -> bf16 weight conversion
// ---------------------------------------------------------------------------
__global__ __launch_bounds__(256) void f2bf_kernel(const float* __restrict__ w,
                                                   bf16* __restrict__ o, int n) {
    int i = blockIdx.x * 256 + threadIdx.x;
    if (i < n) o[i] = (bf16)w[i];
}

// ---------------------------------------------------------------------------
// Fused SageConv GEMM:  H[row, :128] = relu( [Aself(row) | Aneigh(row)] @ W^T )
// (neigh already carries the 1/(deg+1) scaling from the aggregator.)
// W is [128, 2*KS] row-major in bf16, staged whole in LDS. Block = 8 waves;
// each wave computes a 16x128 strip via 8 f32 accumulators; K fully unrolled.
// An empty-asm liveness barrier pins all 8 B fragments in registers so the
// 16 ds_load_b128's issue back-to-back ahead of one s_wait_dscnt and 8
// back-to-back WMMAs (instead of one DS round-trip per WMMA).
// ---------------------------------------------------------------------------
template <int KS>   // per-half K: 256 (layer1) or 128 (layer2); total K = 2*KS
__global__ __launch_bounds__(256)
void gemm_sage(const float* __restrict__ Aself, const float* __restrict__ Aneigh,
               const bf16* __restrict__ Wbf, float* __restrict__ Hout, int nrows) {
    constexpr int KW = 2 * KS;
    extern __shared__ char smem_raw[];
    bf16* sW = (bf16*)smem_raw;              // [128, KW] bf16

    {   // cooperative 16B-granule load of the full weight matrix into LDS
        const uint4* gw = (const uint4*)Wbf;
        uint4* sv = (uint4*)sW;
        const int n16 = KW * 128 * 2 / 16;
        for (int i = threadIdx.x; i < n16; i += 256) sv[i] = gw[i];
    }
    __syncthreads();

    const int wave = threadIdx.x >> 5;
    const int lane = threadIdx.x & 31;
    const int m    = lane & 15;              // tile row / B column
    const int kh   = lane >> 4;              // K-half selector
    const int rowBase = blockIdx.x * 128 + wave * 16;
    const int row  = rowBase + m;
    const int rowc = (row < nrows) ? row : 0;    // clamp for safe loads

    const float* selfRow  = Aself  + (size_t)rowc * KS;
    const float* neighRow = Aneigh + (size_t)rowc * KS;
    const bf16*  sWrow    = sW + (size_t)m * KW + kh * 16;

    v8f acc[8] = {};

    #pragma unroll
    for (int kb = 0; kb < KW; kb += 32) {
        // ---- A fragment: two 8-float groups, f32 -> bf16 on the fly ----
        v16bf afrag;
        #pragma unroll
        for (int g = 0; g < 2; ++g) {
            const int ks = kb + kh * 16 + g * 8;   // compile-time half select
            const float* src = (ks < KS) ? (selfRow + ks) : (neighRow + (ks - KS));
            const float4 f0 = *(const float4*)(src);
            const float4 f1 = *(const float4*)(src + 4);
            afrag[g * 8 + 0] = (bf16)f0.x;
            afrag[g * 8 + 1] = (bf16)f0.y;
            afrag[g * 8 + 2] = (bf16)f0.z;
            afrag[g * 8 + 3] = (bf16)f0.w;
            afrag[g * 8 + 4] = (bf16)f1.x;
            afrag[g * 8 + 5] = (bf16)f1.y;
            afrag[g * 8 + 6] = (bf16)f1.z;
            afrag[g * 8 + 7] = (bf16)f1.w;
        }
        // ---- batch-load the 8 B fragments ----
        v16bf b0 = *(const v16bf*)(sWrow + (size_t)0 * 16 * KW + kb);
        v16bf b1 = *(const v16bf*)(sWrow + (size_t)1 * 16 * KW + kb);
        v16bf b2 = *(const v16bf*)(sWrow + (size_t)2 * 16 * KW + kb);
        v16bf b3 = *(const v16bf*)(sWrow + (size_t)3 * 16 * KW + kb);
        v16bf b4 = *(const v16bf*)(sWrow + (size_t)4 * 16 * KW + kb);
        v16bf b5 = *(const v16bf*)(sWrow + (size_t)5 * 16 * KW + kb);
        v16bf b6 = *(const v16bf*)(sWrow + (size_t)6 * 16 * KW + kb);
        v16bf b7 = *(const v16bf*)(sWrow + (size_t)7 * 16 * KW + kb);
        // liveness barrier: keep all 8 fragments materialized simultaneously
        asm volatile("" : "+v"(b0), "+v"(b1), "+v"(b2), "+v"(b3),
                          "+v"(b4), "+v"(b5), "+v"(b6), "+v"(b7));
        // ---- 8 back-to-back WMMAs ----
        acc[0] = __builtin_amdgcn_wmma_f32_16x16x32_bf16(false, afrag, false, b0,
                                                         (short)0, acc[0], false, false);
        acc[1] = __builtin_amdgcn_wmma_f32_16x16x32_bf16(false, afrag, false, b1,
                                                         (short)0, acc[1], false, false);
        acc[2] = __builtin_amdgcn_wmma_f32_16x16x32_bf16(false, afrag, false, b2,
                                                         (short)0, acc[2], false, false);
        acc[3] = __builtin_amdgcn_wmma_f32_16x16x32_bf16(false, afrag, false, b3,
                                                         (short)0, acc[3], false, false);
        acc[4] = __builtin_amdgcn_wmma_f32_16x16x32_bf16(false, afrag, false, b4,
                                                         (short)0, acc[4], false, false);
        acc[5] = __builtin_amdgcn_wmma_f32_16x16x32_bf16(false, afrag, false, b5,
                                                         (short)0, acc[5], false, false);
        acc[6] = __builtin_amdgcn_wmma_f32_16x16x32_bf16(false, afrag, false, b6,
                                                         (short)0, acc[6], false, false);
        acc[7] = __builtin_amdgcn_wmma_f32_16x16x32_bf16(false, afrag, false, b7,
                                                         (short)0, acc[7], false, false);
    }

    // ---- relu + store; uniform fast path for full tiles ----
    if (rowBase + 16 <= nrows) {
        #pragma unroll
        for (int t = 0; t < 8; ++t) {
            const int col = t * 16 + m;
            #pragma unroll
            for (int r = 0; r < 8; ++r) {
                const int orow = rowBase + kh * 8 + r;
                float v = acc[t][r];
                Hout[(size_t)orow * 128 + col] = v > 0.0f ? v : 0.0f;
            }
        }
    } else {
        #pragma unroll
        for (int t = 0; t < 8; ++t) {
            const int col = t * 16 + m;
            #pragma unroll
            for (int r = 0; r < 8; ++r) {
                const int orow = rowBase + kh * 8 + r;
                if (orow < nrows) {
                    float v = acc[t][r];
                    Hout[(size_t)orow * 128 + col] = v > 0.0f ? v : 0.0f;
                }
            }
        }
    }
}

// ---------------------------------------------------------------------------
// Head: logits = h2 @ mlpW^T + b ; log_softmax over 40 classes. Block/node.
// ---------------------------------------------------------------------------
__global__ __launch_bounds__(64)
void head_kernel(const float* __restrict__ h2, const float* __restrict__ Wc,
                 const float* __restrict__ bc, float* __restrict__ out, int N) {
    __shared__ float sh[128];
    __shared__ float sl[40];
    __shared__ float sred[2];
    const int node = blockIdx.x;
    const int t = threadIdx.x;
    if (node >= N) return;
    sh[t]      = h2[(size_t)node * 128 + t];
    sh[t + 64] = h2[(size_t)node * 128 + t + 64];
    __syncthreads();
    if (t < 40) {
        const float* w = Wc + t * 128;
        float acc = bc[t];
        #pragma unroll 4
        for (int k = 0; k < 128; ++k) acc += sh[k] * w[k];
        sl[t] = acc;
    }
    __syncthreads();
    if (t == 0) {
        float mx = sl[0];
        for (int c = 1; c < 40; ++c) mx = fmaxf(mx, sl[c]);
        float s = 0.0f;
        for (int c = 0; c < 40; ++c) s += expf(sl[c] - mx);
        sred[0] = mx;
        sred[1] = logf(s);
    }
    __syncthreads();
    if (t < 40) out[(size_t)node * 40 + t] = sl[t] - sred[0] - sred[1];
}

// ---------------------------------------------------------------------------
// launch
// ---------------------------------------------------------------------------
extern "C" void kernel_launch(void* const* d_in, const int* in_sizes, int n_in,
                              void* d_out, int out_size, void* d_ws, size_t ws_size,
                              hipStream_t stream) {
    const float* x    = (const float*)d_in[0];   // [N, 256]
    const float* W1   = (const float*)d_in[1];   // [128, 512]
    const float* W2   = (const float*)d_in[2];   // [128, 256]
    const float* mlpW = (const float*)d_in[3];   // [40, 128]
    const float* mlpb = (const float*)d_in[4];   // [40]
    const int*   rows = (const int*)d_in[5];     // [E]
    const int*   cols = (const int*)d_in[6];     // [E]
    const int N = in_sizes[0] / 256;
    const int E = in_sizes[5];
    const int nb = (N + 1023) / 1024;            // scan tiles

    // workspace carve-out (256B aligned)
    char* ws = (char*)d_ws;
    size_t off = 0;
    auto carve = [&](size_t bytes) -> void* {
        off = (off + 255) & ~(size_t)255;
        void* p = ws + off;
        off += bytes;
        return p;
    };
    int*   cnt      = (int*)carve((size_t)N * 4);
    int*   rowstart = (int*)carve((size_t)N * 4);
    int*   cur      = (int*)carve((size_t)N * 4);
    int*   blockSums= (int*)carve((size_t)nb * 4);
    int*   adj      = (int*)carve((size_t)E * 4);
    float* inv      = (float*)carve((size_t)N * 4);
    float* neigh    = (float*)carve((size_t)N * 256 * 4);   // layer1; reused layer2
    float* h1       = (float*)carve((size_t)N * 128 * 4);
    float* h2       = (float*)carve((size_t)N * 128 * 4);
    bf16*  W1b      = (bf16*)carve((size_t)512 * 128 * 2);
    bf16*  W2b      = (bf16*)carve((size_t)256 * 128 * 2);

    // ---- degree + CSR build ----
    hipMemsetAsync(cnt, 0, (size_t)N * 4, stream);
    hipMemsetAsync(cur, 0, (size_t)N * 4, stream);
    count_kernel<<<(E + 255) / 256, 256, 0, stream>>>(rows, cnt, E);
    inv_kernel<<<(N + 255) / 256, 256, 0, stream>>>(cnt, inv, N);
    scan_partial<<<nb, 256, 0, stream>>>(cnt, blockSums, N);
    scan_sums<<<1, 1, 0, stream>>>(blockSums, nb);
    scan_final<<<nb, 256, 0, stream>>>(cnt, blockSums, rowstart, N);
    fill_adj<<<(E + 255) / 256, 256, 0, stream>>>(rows, cols, rowstart, cur, adj, E);

    // ---- bf16 weights ----
    f2bf_kernel<<<(512 * 128 + 255) / 256, 256, 0, stream>>>(W1, W1b, 512 * 128);
    f2bf_kernel<<<(256 * 128 + 255) / 256, 256, 0, stream>>>(W2, W2b, 256 * 128);

    // ---- layer 1: gather-aggregate + fused concat-GEMM + relu ----
    aggregate<256><<<(N + 7) / 8, 256, 0, stream>>>(x, adj, rowstart, cnt, inv,
                                                    neigh, N);
    gemm_sage<256><<<(N + 127) / 128, 256, 512 * 128 * 2, stream>>>(
        x, neigh, W1b, h1, N);

    // ---- layer 2 ----
    aggregate<128><<<(N + 7) / 8, 256, 0, stream>>>(h1, adj, rowstart, cnt, inv,
                                                    neigh, N);
    gemm_sage<128><<<(N + 127) / 128, 256, 256 * 128 * 2, stream>>>(
        h1, neigh, W2b, h2, N);

    // ---- head + log_softmax ----
    head_kernel<<<N, 64, 0, stream>>>(h2, mlpW, mlpb, (float*)d_out, N);
}
